// EGNNLayer_31791347925392
// MI455X (gfx1250) — compile-verified, matrix operand order checked by reading, not run on hardware
//
#include <hip/hip_runtime.h>
#include <math.h>

typedef _Float16 h16;
typedef __attribute__((ext_vector_type(16))) _Float16 v16h;
typedef __attribute__((ext_vector_type(8)))  _Float16 v8h;
typedef __attribute__((ext_vector_type(8)))  float    v8f;

#define NODE_DIM 128
#define HIDDEN   128
#define NBASIS   32
#define MSG_IN   288      // 2*NODE_DIM + NBASIS
#define WAVES    4        // waves per workgroup

static __device__ __forceinline__ v16h cat8(v8h a, v8h b) {
  return __builtin_shufflevector(a, b, 0,1,2,3,4,5,6,7,8,9,10,11,12,13,14,15);
}
static __device__ __forceinline__ float silu(float x) {
  return x / (1.0f + __expf(-x));
}
// B fragment: pre-swizzled by pack_w so each lane reads 32 contiguous bytes.
static __device__ __forceinline__ v16h load_b_frag(const h16* __restrict__ wp,
                                                   int kt, int nt, int NT, int lane) {
  const v8h* p = (const v8h*)(wp + (size_t)(((kt * NT + nt) * 32 + lane)) * 16);
  return cat8(p[0], p[1]);
}
// A fragment from a row-major f16 LDS panel (row stride in halves = rs).
// lane<16: K = kb+0..7 / kb+16..23 ; lane>=16: kb=8 offsets (ISA 16-bit A layout).
static __device__ __forceinline__ v16h load_a_frag(const h16* row_base, int kt, int kb) {
  const v8h* a0 = (const v8h*)(row_base + kt * 32 + kb);
  const v8h* a1 = (const v8h*)(row_base + kt * 32 + 16 + kb);
  return cat8(*a0, *a1);
}

// -------- prep: swizzle f32 weights [K][N] into WMMA B-fragment order (f16) -----
extern "C" __global__ void pack_w(const float* __restrict__ W, h16* __restrict__ out,
                                  int K, int Nn) {
  int t = blockIdx.x * blockDim.x + threadIdx.x;
  int KT = K >> 5, NT = Nn >> 4;
  int total = KT * NT * 32;
  if (t >= total) return;
  int lane = t & 31;
  int tile = t >> 5;
  int nt = tile % NT, kt = tile / NT;
  int n = nt * 16 + (lane & 15);
  int kbase = (lane < 16) ? 0 : 8;
  alignas(16) h16 tmp[16];
#pragma unroll
  for (int r = 0; r < 4; ++r) {
    int k0 = kt * 32 + kbase + 2 * r;          // VGPRs 0..3
    tmp[2 * r]     = (h16)W[(size_t)k0 * Nn + n];
    tmp[2 * r + 1] = (h16)W[(size_t)(k0 + 1) * Nn + n];
    int k1 = kt * 32 + 16 + kbase + 2 * r;     // VGPRs 4..7
    tmp[8 + 2 * r]     = (h16)W[(size_t)k1 * Nn + n];
    tmp[8 + 2 * r + 1] = (h16)W[(size_t)(k1 + 1) * Nn + n];
  }
  v8h* o = (v8h*)(out + (size_t)t * 16);
  o[0] = *(const v8h*)(tmp);
  o[1] = *(const v8h*)(tmp + 8);
}

// -------- init: zero aggr, pos_out = pos ---------------------------------------
extern "C" __global__ void egnn_init(float* __restrict__ aggr, const float* __restrict__ pos,
                                     float* __restrict__ pos_out, int N) {
  int i = blockIdx.x * blockDim.x + threadIdx.x;
  int stride = gridDim.x * blockDim.x;
  for (int j = i; j < N * HIDDEN; j += stride) aggr[j] = 0.0f;
  for (int j = i; j < N * 3; j += stride) pos_out[j] = pos[j];
}

// -------- edge kernel: message MLP + coord MLP + segment-sum atomics -----------
extern "C" __global__ void __launch_bounds__(WAVES * 32)
egnn_edge(const float* __restrict__ hfeat, const float* __restrict__ pos,
          const int* __restrict__ ei,
          const h16* __restrict__ mw1p, const float* __restrict__ mb1,
          const h16* __restrict__ mw2p, const float* __restrict__ mb2,
          const h16* __restrict__ cw1p, const float* __restrict__ cb1,
          const float* __restrict__ cw2, const float* __restrict__ cb2,
          float* __restrict__ aggr, float* __restrict__ pos_out, int E) {
  extern __shared__ char smem[];
  const int wave = threadIdx.x >> 5;
  const int lane = threadIdx.x & 31;
  const int m    = lane & 15;                 // edge row within tile / matrix col
  const int kb   = (lane < 16) ? 0 : 8;       // A/B fragment K sub-offset
  const int drow = (lane < 16) ? 0 : 8;       // D-layout row base

  const int WB = 16 * MSG_IN * 2 + 16 * HIDDEN * 2 + 64 + 64;
  char* wb = smem + wave * WB;
  h16*   s_min = (h16*)wb;                              // [16][288] f16
  h16*   s_hid = (h16*)(wb + 16 * MSG_IN * 2);          // [16][128] f16
  int*   s_row = (int*)(wb + 16 * MSG_IN * 2 + 16 * HIDDEN * 2);
  float* s_cm  = (float*)(wb + 16 * MSG_IN * 2 + 16 * HIDDEN * 2 + 64);

  const int ntile = E >> 4;
  const int nwav  = gridDim.x * WAVES;
  for (int tile = blockIdx.x * WAVES + wave; tile < ntile; tile += nwav) {
    const int e    = tile * 16 + m;
    const int vrow = ei[e];
    const int vcol = ei[E + e];
    float rx = 0.f, ry = 0.f, rz = 0.f;

    // ---- stage msg_in = [h[row] | h[col] | gauss(dist)] as f16 in LDS --------
    if (lane < 16) {
      s_row[m] = vrow;
      rx = pos[vrow * 3 + 0] - pos[vcol * 3 + 0];
      ry = pos[vrow * 3 + 1] - pos[vcol * 3 + 1];
      rz = pos[vrow * 3 + 2] - pos[vcol * 3 + 2];
      float dist = sqrtf(rx * rx + ry * ry + rz * rz + 1e-8f);
      const float step  = 10.0f / (NBASIS - 1);
      const float coeff = -0.5f / (step * step);
#pragma unroll
      for (int b = 0; b < NBASIS; ++b) {
        float d = dist - step * b;
        s_min[m * MSG_IN + 2 * NODE_DIM + b] = (h16)__expf(coeff * d * d);
      }
      const float4* hp = (const float4*)(hfeat + (size_t)vrow * NODE_DIM);
#pragma unroll 4
      for (int j = 0; j < NODE_DIM / 4; ++j) {
        float4 v = hp[j];
        h16* dst = s_min + m * MSG_IN + j * 4;
        dst[0] = (h16)v.x; dst[1] = (h16)v.y; dst[2] = (h16)v.z; dst[3] = (h16)v.w;
      }
    } else {
      const float4* hp = (const float4*)(hfeat + (size_t)vcol * NODE_DIM);
#pragma unroll 4
      for (int j = 0; j < NODE_DIM / 4; ++j) {
        float4 v = hp[j];
        h16* dst = s_min + m * MSG_IN + NODE_DIM + j * 4;
        dst[0] = (h16)v.x; dst[1] = (h16)v.y; dst[2] = (h16)v.z; dst[3] = (h16)v.w;
      }
    }
    asm volatile("s_wait_dscnt 0" ::: "memory");

    // ---- message MLP layer 1: [16x288] x [288x128], SiLU --------------------
    v8f acc[8];
#pragma unroll
    for (int nt = 0; nt < 8; ++nt) {
      float b0 = mb1[nt * 16 + m];
      acc[nt] = (v8f){b0, b0, b0, b0, b0, b0, b0, b0};
    }
#pragma unroll
    for (int kt = 0; kt < MSG_IN / 32; ++kt) {
      v16h a = load_a_frag(s_min + m * MSG_IN, kt, kb);
#pragma unroll
      for (int nt = 0; nt < 8; ++nt) {
        v16h b = load_b_frag(mw1p, kt, nt, 8, lane);
        acc[nt] = __builtin_amdgcn_wmma_f32_16x16x32_f16(false, a, false, b,
                                                         (short)0, acc[nt], false, false);
      }
    }
#pragma unroll
    for (int nt = 0; nt < 8; ++nt)
#pragma unroll
      for (int r = 0; r < 8; ++r)
        s_hid[(drow + r) * HIDDEN + nt * 16 + m] = (h16)silu(acc[nt][r]);
    asm volatile("s_wait_dscnt 0" ::: "memory");

    // ---- message MLP layer 2: [16x128] x [128x128] --------------------------
    v8f acc2[8];
#pragma unroll
    for (int nt = 0; nt < 8; ++nt) {
      float b0 = mb2[nt * 16 + m];
      acc2[nt] = (v8f){b0, b0, b0, b0, b0, b0, b0, b0};
    }
#pragma unroll
    for (int kt = 0; kt < HIDDEN / 32; ++kt) {
      v16h a = load_a_frag(s_hid + m * HIDDEN, kt, kb);
#pragma unroll
      for (int nt = 0; nt < 8; ++nt) {
        v16h b = load_b_frag(mw2p, kt, nt, 8, lane);
        acc2[nt] = __builtin_amdgcn_wmma_f32_16x16x32_f16(false, a, false, b,
                                                          (short)0, acc2[nt], false, false);
      }
    }
    // messages -> segment_sum atomics + restage as f16 A-panel for coord MLP
#pragma unroll
    for (int nt = 0; nt < 8; ++nt) {
      const int col = nt * 16 + m;
#pragma unroll
      for (int r = 0; r < 8; ++r) {
        float v  = acc2[nt][r];
        int   rr = drow + r;
        unsafeAtomicAdd(&aggr[(size_t)s_row[rr] * HIDDEN + col], v);
        s_hid[rr * HIDDEN + col] = (h16)v;
      }
    }
    asm volatile("s_wait_dscnt 0" ::: "memory");

    // ---- coord MLP: silu(msg@cw1+cb1) then dot with cw2 ---------------------
    v8f acc3[8];
#pragma unroll
    for (int nt = 0; nt < 8; ++nt) {
      float b0 = cb1[nt * 16 + m];
      acc3[nt] = (v8f){b0, b0, b0, b0, b0, b0, b0, b0};
    }
#pragma unroll
    for (int kt = 0; kt < HIDDEN / 32; ++kt) {
      v16h a = load_a_frag(s_hid + m * HIDDEN, kt, kb);
#pragma unroll
      for (int nt = 0; nt < 8; ++nt) {
        v16h b = load_b_frag(cw1p, kt, nt, 8, lane);
        acc3[nt] = __builtin_amdgcn_wmma_f32_16x16x32_f16(false, a, false, b,
                                                          (short)0, acc3[nt], false, false);
      }
    }
    if (lane < 16) s_cm[m] = cb2[0];
    asm volatile("s_wait_dscnt 0" ::: "memory");
#pragma unroll
    for (int nt = 0; nt < 8; ++nt) {
      float w = cw2[nt * 16 + m];
#pragma unroll
      for (int r = 0; r < 8; ++r)
        atomicAdd(&s_cm[drow + r], silu(acc3[nt][r]) * w);   // ds_add_f32
    }
    asm volatile("s_wait_dscnt 0" ::: "memory");
    if (lane < 16) {
      float cm = s_cm[m];
      unsafeAtomicAdd(&pos_out[vrow * 3 + 0], rx * cm);
      unsafeAtomicAdd(&pos_out[vrow * 3 + 1], ry * cm);
      unsafeAtomicAdd(&pos_out[vrow * 3 + 2], rz * cm);
    }
    asm volatile("s_wait_dscnt 0" ::: "memory");
  }
}

// -------- node kernel: h_out = h + MLP([h | aggr]) ----------------------------
extern "C" __global__ void __launch_bounds__(WAVES * 32)
egnn_node(const float* __restrict__ hfeat, const float* __restrict__ aggr,
          const h16* __restrict__ nw1p, const float* __restrict__ nb1,
          const h16* __restrict__ nw2p, const float* __restrict__ nb2,
          float* __restrict__ h_out, int Nn) {
  extern __shared__ char smem[];
  const int wave = threadIdx.x >> 5;
  const int lane = threadIdx.x & 31;
  const int m    = lane & 15;
  const int kb   = (lane < 16) ? 0 : 8;
  const int drow = (lane < 16) ? 0 : 8;

  const int NIN = 2 * NODE_DIM;              // 256
  const int WB  = 16 * NIN * 2 + 16 * HIDDEN * 2;
  char* wb = smem + wave * WB;
  h16* s_nin = (h16*)wb;                     // [16][256] f16
  h16* s_hid = (h16*)(wb + 16 * NIN * 2);    // [16][128] f16

  const int ntile = Nn >> 4;
  const int nwav  = gridDim.x * WAVES;
  for (int tile = blockIdx.x * WAVES + wave; tile < ntile; tile += nwav) {
    const int node = tile * 16 + m;
    const float4* src = (lane < 16) ? (const float4*)(hfeat + (size_t)node * NODE_DIM)
                                    : (const float4*)(aggr + (size_t)node * HIDDEN);
    h16* dst = s_nin + m * NIN + ((lane < 16) ? 0 : NODE_DIM);
#pragma unroll 4
    for (int j = 0; j < NODE_DIM / 4; ++j) {
      float4 v = src[j];
      dst[j * 4 + 0] = (h16)v.x; dst[j * 4 + 1] = (h16)v.y;
      dst[j * 4 + 2] = (h16)v.z; dst[j * 4 + 3] = (h16)v.w;
    }
    asm volatile("s_wait_dscnt 0" ::: "memory");

    v8f acc[8];
#pragma unroll
    for (int nt = 0; nt < 8; ++nt) {
      float b0 = nb1[nt * 16 + m];
      acc[nt] = (v8f){b0, b0, b0, b0, b0, b0, b0, b0};
    }
#pragma unroll
    for (int kt = 0; kt < NIN / 32; ++kt) {
      v16h a = load_a_frag(s_nin + m * NIN, kt, kb);
#pragma unroll
      for (int nt = 0; nt < 8; ++nt) {
        v16h b = load_b_frag(nw1p, kt, nt, 8, lane);
        acc[nt] = __builtin_amdgcn_wmma_f32_16x16x32_f16(false, a, false, b,
                                                         (short)0, acc[nt], false, false);
      }
    }
#pragma unroll
    for (int nt = 0; nt < 8; ++nt)
#pragma unroll
      for (int r = 0; r < 8; ++r)
        s_hid[(drow + r) * HIDDEN + nt * 16 + m] = (h16)silu(acc[nt][r]);
    asm volatile("s_wait_dscnt 0" ::: "memory");

    v8f acc2[8];
#pragma unroll
    for (int nt = 0; nt < 8; ++nt) {
      float b0 = nb2[nt * 16 + m];
      acc2[nt] = (v8f){b0, b0, b0, b0, b0, b0, b0, b0};
    }
#pragma unroll
    for (int kt = 0; kt < HIDDEN / 32; ++kt) {
      v16h a = load_a_frag(s_hid + m * HIDDEN, kt, kb);
#pragma unroll
      for (int nt = 0; nt < 8; ++nt) {
        v16h b = load_b_frag(nw2p, kt, nt, 8, lane);
        acc2[nt] = __builtin_amdgcn_wmma_f32_16x16x32_f16(false, a, false, b,
                                                          (short)0, acc2[nt], false, false);
      }
    }
#pragma unroll
    for (int nt = 0; nt < 8; ++nt) {
      const int col = nt * 16 + m;
#pragma unroll
      for (int r = 0; r < 8; ++r) {
        int n2 = tile * 16 + drow + r;
        h_out[(size_t)n2 * NODE_DIM + col] =
            hfeat[(size_t)n2 * NODE_DIM + col] + acc2[nt][r];
      }
    }
  }
}

extern "C" void kernel_launch(void* const* d_in, const int* in_sizes, int n_in,
                              void* d_out, int out_size, void* d_ws, size_t ws_size,
                              hipStream_t stream) {
  const float* h   = (const float*)d_in[0];
  const float* pos = (const float*)d_in[1];
  const int*   ei  = (const int*)d_in[2];
  const float* mw1 = (const float*)d_in[3];  const float* mb1 = (const float*)d_in[4];
  const float* mw2 = (const float*)d_in[5];  const float* mb2 = (const float*)d_in[6];
  const float* nw1 = (const float*)d_in[7];  const float* nb1 = (const float*)d_in[8];
  const float* nw2 = (const float*)d_in[9];  const float* nb2 = (const float*)d_in[10];
  const float* cw1 = (const float*)d_in[11]; const float* cb1 = (const float*)d_in[12];
  const float* cw2 = (const float*)d_in[13]; const float* cb2 = (const float*)d_in[14];
  const int N = in_sizes[0] / NODE_DIM;      // 50000
  const int E = in_sizes[2] / 2;             // 800000

  // workspace layout: packed f16 weights, then f32 aggr buffer
  char* ws = (char*)d_ws;
  h16* mw1p = (h16*)(ws + 0);        // 288*128 f16 = 73728 B
  h16* mw2p = (h16*)(ws + 73728);    // 128*128 f16 = 32768 B
  h16* cw1p = (h16*)(ws + 106496);   // 32768 B
  h16* nw1p = (h16*)(ws + 139264);   // 256*128 f16 = 65536 B
  h16* nw2p = (h16*)(ws + 204800);   // 32768 B
  float* aggr = (float*)(ws + 237568);
  float* h_out   = (float*)d_out;
  float* pos_out = h_out + (size_t)N * NODE_DIM;

  // re-swizzle weights each call (deterministic, graph-capture safe)
  pack_w<<<(9 * 8 * 32 + 127) / 128, 128, 0, stream>>>(mw1, mw1p, MSG_IN, HIDDEN);
  pack_w<<<(4 * 8 * 32 + 127) / 128, 128, 0, stream>>>(mw2, mw2p, HIDDEN, HIDDEN);
  pack_w<<<(4 * 8 * 32 + 127) / 128, 128, 0, stream>>>(cw1, cw1p, HIDDEN, HIDDEN);
  pack_w<<<(8 * 8 * 32 + 127) / 128, 128, 0, stream>>>(nw1, nw1p, 2 * NODE_DIM, HIDDEN);
  pack_w<<<(4 * 8 * 32 + 127) / 128, 128, 0, stream>>>(nw2, nw2p, HIDDEN, NODE_DIM);
  egnn_init<<<2048, 256, 0, stream>>>(aggr, pos, pos_out, N);

  size_t lds_e = (size_t)WAVES * (16 * MSG_IN * 2 + 16 * HIDDEN * 2 + 64 + 64);
  egnn_edge<<<2048, WAVES * 32, lds_e, stream>>>(h, pos, ei, mw1p, mb1, mw2p, mb2,
                                                 cw1p, cb1, cw2, cb2, aggr, pos_out, E);

  size_t lds_n = (size_t)WAVES * (16 * 2 * NODE_DIM * 2 + 16 * HIDDEN * 2);
  egnn_node<<<800, WAVES * 32, lds_n, stream>>>(h, aggr, nw1p, nb1, nw2p, nb2, h_out, N);
}
